// TARigNet_24816321036338
// MI455X (gfx1250) — matrix operand order repaired and checked by reading, not compiled, not run on hardware
//
#include <hip/hip_runtime.h>
#include <hip/hip_bf16.h>
#include <math.h>

// ---------------------------------------------------------------------------
// Types for CDNA5 WMMA (wave32): v_wmma_f32_16x16x32_f16
// ---------------------------------------------------------------------------
typedef _Float16 h8v  __attribute__((ext_vector_type(8)));
typedef _Float16 v16h __attribute__((ext_vector_type(16)));
typedef float    v8f  __attribute__((ext_vector_type(8)));

union ABfrag { v16h v; h8v h[2]; };

// A fragment (16-bit A 16x32 layout):
//   lanes 0-15 : row=l16, elems 0-7 = K 0..7,  elems 8-15 = K 8..15
//   lanes 16-31: row=l16, elems 0-7 = K 16..23, elems 8-15 = K 24..31
// B fragment (16-bit B 32x16): lanes 0-15 col=l16 K 0..15; lanes 16-31 K 16..31
// C/D layout: vgpr r -> M = r + 8*hi, N = l16.

// ---------------------------------------------------------------------------
// Single-tile WMMA GEMM (fallback for N=32): wave -> one 16x16 tile.
// Block = 8 waves covering 64(M) x 32(N). grid = (N/32, M/64).
// ---------------------------------------------------------------------------
__global__ __launch_bounds__(256) void k_gemm_wmma(
    const _Float16* __restrict__ A, int lda,
    const _Float16* __restrict__ Wt,
    const float* __restrict__ bias,
    _Float16* __restrict__ outH, float* __restrict__ outF, int ldo,
    int M, int N, int K, int relu)
{
  const int lane = threadIdx.x & 31;
  const int wave = threadIdx.x >> 5;
  const int tm = wave >> 1, tn = wave & 1;
  const int row0 = blockIdx.y * 64 + tm * 16;
  const int col0 = blockIdx.x * 32 + tn * 16;
  const int l16 = lane & 15;
  const int hi  = lane >> 4;

  const _Float16* ap = A  + (size_t)(row0 + l16) * (size_t)lda + hi * 16;
  const _Float16* bp = Wt + (size_t)(col0 + l16) * (size_t)K  + hi * 16;

  v8f acc = {};
  for (int k = 0; k < K; k += 32) {
    ABfrag a;
    a.h[0] = *(const h8v*)(ap + k);
    a.h[1] = *(const h8v*)(ap + k + 8);
    v16h b = *(const v16h*)(bp + k);
    __builtin_prefetch(ap + k + 64, 0, 1);
    __builtin_prefetch(bp + k + 64, 0, 1);
    acc = __builtin_amdgcn_wmma_f32_16x16x32_f16(
        false, a.v, false, b, (short)0, acc, false, false);
  }

  const int col = col0 + l16;
  const float bv = bias ? bias[col] : 0.0f;
#pragma unroll
  for (int r = 0; r < 8; ++r) {
    int row = row0 + r + hi * 8;
    float v = acc[r] + bv;
    if (relu) v = v > 0.0f ? v : 0.0f;
    size_t o = (size_t)row * (size_t)ldo + col;
    if (outH) outH[o] = (_Float16)v;
    if (outF) outF[o] = v;
  }
}

// ---------------------------------------------------------------------------
// 2x2-tiled WMMA GEMM: wave -> 32x32 output (4 accumulators), halving the
// load:wmma ratio (8 b128 loads -> 4 wmma per k-step) and doubling register
// reuse of A/B fragments. Block = 8 waves (4x2) covering 128(M) x 64(N).
// grid = (N/64, M/128). Requires M%128==0, N%64==0.
// ---------------------------------------------------------------------------
__global__ __launch_bounds__(256) void k_gemm_wmma_2x2(
    const _Float16* __restrict__ A, int lda,
    const _Float16* __restrict__ Wt,
    const float* __restrict__ bias,
    _Float16* __restrict__ outH, float* __restrict__ outF, int ldo,
    int M, int N, int K, int relu)
{
  const int lane = threadIdx.x & 31;
  const int wave = threadIdx.x >> 5;
  const int wy = wave >> 1, wx = wave & 1;
  const int row0 = blockIdx.y * 128 + wy * 32;
  const int col0 = blockIdx.x * 64 + wx * 32;
  const int l16 = lane & 15;
  const int hi  = lane >> 4;

  const _Float16* ap0 = A  + (size_t)(row0 + l16) * (size_t)lda + hi * 16;
  const _Float16* ap1 = ap0 + (size_t)16 * lda;
  const _Float16* bp0 = Wt + (size_t)(col0 + l16) * (size_t)K  + hi * 16;
  const _Float16* bp1 = bp0 + (size_t)16 * K;

  v8f acc00 = {}, acc01 = {}, acc10 = {}, acc11 = {};
  for (int k = 0; k < K; k += 32) {
    ABfrag a0, a1;
    a0.h[0] = *(const h8v*)(ap0 + k);
    a0.h[1] = *(const h8v*)(ap0 + k + 8);
    a1.h[0] = *(const h8v*)(ap1 + k);
    a1.h[1] = *(const h8v*)(ap1 + k + 8);
    v16h b0 = *(const v16h*)(bp0 + k);
    v16h b1 = *(const v16h*)(bp1 + k);
    __builtin_prefetch(ap0 + k + 64, 0, 1);
    __builtin_prefetch(ap1 + k + 64, 0, 1);
    __builtin_prefetch(bp0 + k + 64, 0, 1);
    __builtin_prefetch(bp1 + k + 64, 0, 1);
    acc00 = __builtin_amdgcn_wmma_f32_16x16x32_f16(false, a0.v, false, b0, (short)0, acc00, false, false);
    acc01 = __builtin_amdgcn_wmma_f32_16x16x32_f16(false, a0.v, false, b1, (short)0, acc01, false, false);
    acc10 = __builtin_amdgcn_wmma_f32_16x16x32_f16(false, a1.v, false, b0, (short)0, acc10, false, false);
    acc11 = __builtin_amdgcn_wmma_f32_16x16x32_f16(false, a1.v, false, b1, (short)0, acc11, false, false);
  }

  const int c0 = col0 + l16, c1 = c0 + 16;
  const float bv0 = bias ? bias[c0] : 0.0f;
  const float bv1 = bias ? bias[c1] : 0.0f;
#pragma unroll
  for (int r = 0; r < 8; ++r) {
    int rA = row0 + r + hi * 8;
    int rB = rA + 16;
    float v00 = acc00[r] + bv0, v01 = acc01[r] + bv1;
    float v10 = acc10[r] + bv0, v11 = acc11[r] + bv1;
    if (relu) {
      v00 = v00 > 0.0f ? v00 : 0.0f;  v01 = v01 > 0.0f ? v01 : 0.0f;
      v10 = v10 > 0.0f ? v10 : 0.0f;  v11 = v11 > 0.0f ? v11 : 0.0f;
    }
    size_t oA = (size_t)rA * (size_t)ldo;
    size_t oB = (size_t)rB * (size_t)ldo;
    if (outH) {
      outH[oA + c0] = (_Float16)v00;  outH[oA + c1] = (_Float16)v01;
      outH[oB + c0] = (_Float16)v10;  outH[oB + c1] = (_Float16)v11;
    }
    if (outF) {
      outF[oA + c0] = v00;  outF[oA + c1] = v01;
      outF[oB + c0] = v10;  outF[oB + c1] = v11;
    }
  }
}

// ---------------------------------------------------------------------------
// Small utility kernels
// ---------------------------------------------------------------------------
__global__ void k_zero_f32(float* __restrict__ p, long n) {
  long i = (long)blockIdx.x * blockDim.x + threadIdx.x;
  if (i < n) p[i] = 0.0f;
}

// W (K x N, fp32) -> Wt (N x Kp, fp16), zero padded rows k >= K
__global__ void k_convw(const float* __restrict__ W, _Float16* __restrict__ Wt,
                        int K, int N, int Kp, long total) {
  long i = (long)blockIdx.x * blockDim.x + threadIdx.x;
  if (i >= total) return;
  int k = (int)(i % Kp);
  int n = (int)(i / Kp);
  Wt[i] = (k < K) ? (_Float16)W[(size_t)k * N + n] : (_Float16)0.0f;
}

__global__ void k_f32_to_f16(const float* __restrict__ in, _Float16* __restrict__ out, long n) {
  long i = (long)blockIdx.x * blockDim.x + threadIdx.x;
  if (i < n) out[i] = (_Float16)in[i];
}

// x0 = concat(pos, normals) padded to width 8 (fp16)
__global__ void k_build_x0(const float* __restrict__ pos, const float* __restrict__ nrm,
                           _Float16* __restrict__ x0, long total) {
  long i = (long)blockIdx.x * blockDim.x + threadIdx.x;
  if (i >= total) return;
  int j = (int)(i % 8); long r = i / 8;
  float v = 0.0f;
  if (j < 3)      v = pos[(size_t)r * 3 + j];
  else if (j < 6) v = nrm[(size_t)r * 3 + (j - 3)];
  x0[i] = (_Float16)v;
}

// ef[e] = [ h[dst], h[src]-h[dst] ] zero padded to lde
__global__ void k_edge_gather(const _Float16* __restrict__ h, int ldh, int ci,
                              const int* __restrict__ src, const int* __restrict__ dst,
                              _Float16* __restrict__ ef, int lde, long total) {
  long i = (long)blockIdx.x * blockDim.x + threadIdx.x;
  if (i >= total) return;
  int j = (int)(i % lde); long e = i / lde;
  _Float16 v = (_Float16)0.0f;
  if (j < ci) {
    v = h[(size_t)dst[e] * ldh + j];
  } else if (j < 2 * ci) {
    int jj = j - ci;
    v = (_Float16)((float)h[(size_t)src[e] * ldh + jj] -
                   (float)h[(size_t)dst[e] * ldh + jj]);
  }
  ef[i] = v;
}

// segment-max of non-negative (post-ReLU) fp16 messages via uint atomicMax
__global__ void k_scatter_max_h(const _Float16* __restrict__ m, int C,
                                const int* __restrict__ dst,
                                float* __restrict__ agg, int ldagg, int colOff,
                                long total) {
  long i = (long)blockIdx.x * blockDim.x + threadIdx.x;
  if (i >= total) return;
  int c = (int)(i % C); long e = i / C;
  float v = (float)m[i];                      // >= 0 (post-ReLU)
  atomicMax((unsigned int*)&agg[(size_t)dst[e] * ldagg + colOff + c],
            __float_as_uint(v));
}

// batch-level segment max of non-negative fp32 (post-ReLU glb output)
__global__ void k_scatter_max_f32(const float* __restrict__ x,
                                  const int* __restrict__ idx,
                                  float* __restrict__ xg, long total) {
  long i = (long)blockIdx.x * blockDim.x + threadIdx.x;
  if (i >= total) return;
  int c = (int)(i % 1024); long r = i / 1024;
  atomicMax((unsigned int*)&xg[(size_t)idx[r] * 1024 + c], __float_as_uint(x[i]));
}

// v = [ xg[batch], x(6), f1(64), f2(256), f3(512), pad->1888 ]  (fp16)
__global__ void k_build_v(const _Float16* __restrict__ x0,
                          const _Float16* __restrict__ feats,
                          const float* __restrict__ xg,
                          const int* __restrict__ batch,
                          _Float16* __restrict__ v, long total) {
  long i = (long)blockIdx.x * blockDim.x + threadIdx.x;
  if (i >= total) return;
  int c = (int)(i % 1888); long r = i / 1888;
  _Float16 val = (_Float16)0.0f;
  if (c < 1024)      val = (_Float16)xg[(size_t)batch[r] * 1024 + c];
  else if (c < 1030) val = x0[(size_t)r * 8 + (c - 1024)];
  else if (c < 1862) val = feats[(size_t)r * 832 + (c - 1030)];
  v[i] = val;
}

__global__ void k_init_small(float* __restrict__ denom, float* __restrict__ coords) {
  int t = threadIdx.x;
  if (t < 96)  denom[t]  = 1e-5f;
  if (t < 288) coords[t] = 0.0f;
}

// heat head tail: z = t@Wo + bo; hm = sigmoid(z); denom += hm[:, :24]
__global__ void k_head_heat(const _Float16* __restrict__ t, const float* __restrict__ Wo,
                            const float* __restrict__ bo, const int* __restrict__ batch,
                            float* __restrict__ hm, float* __restrict__ denom, long total) {
  long i = (long)blockIdx.x * blockDim.x + threadIdx.x;
  if (i >= total) return;
  int c = (int)(i % 48); long r = i / 48;
  const _Float16* tr = t + (size_t)r * 256;
  float s = bo[c];
  for (int k = 0; k < 256; ++k) s += (float)tr[k] * Wo[(size_t)k * 48 + c];
  float sig = 1.0f / (1.0f + __expf(-s));
  hm[i] = sig;
  if (c < 24) atomicAdd(&denom[batch[r] * 24 + c], sig);
}

// coords[b,j,:] += (hm[r,j]/denom[b,j]) * pos[r,:]
__global__ void k_coords(const float* __restrict__ hm, const float* __restrict__ denom,
                         const float* __restrict__ pos, const int* __restrict__ batch,
                         float* __restrict__ coords, long total) {
  long i = (long)blockIdx.x * blockDim.x + threadIdx.x;
  if (i >= total) return;
  int j = (int)(i % 24); long r = i / 24;
  int b = batch[r];
  float w = hm[(size_t)r * 48 + j] / denom[b * 24 + j];
#pragma unroll
  for (int a = 0; a < 3; ++a)
    atomicAdd(&coords[((size_t)b * 24 + j) * 3 + a], w * pos[(size_t)r * 3 + a]);
}

// generic small linear tail (skin head): out = t@Wo + bo
__global__ void k_linear_out(const _Float16* __restrict__ t, const float* __restrict__ Wo,
                             const float* __restrict__ bo, float* __restrict__ out,
                             int C, long total) {
  long i = (long)blockIdx.x * blockDim.x + threadIdx.x;
  if (i >= total) return;
  int c = (int)(i % C); long r = i / C;
  const _Float16* tr = t + (size_t)r * 256;
  float s = bo[c];
  for (int k = 0; k < 256; ++k) s += (float)tr[k] * Wo[(size_t)k * C + c];
  out[i] = s;
}

// conf head tail: 3 dots + L2 normalize with eps
__global__ void k_conf(const _Float16* __restrict__ t, const float* __restrict__ Wo,
                       const float* __restrict__ bo, float* __restrict__ out, int N) {
  int r = blockIdx.x * blockDim.x + threadIdx.x;
  if (r >= N) return;
  const _Float16* tr = t + (size_t)r * 256;
  float z0 = bo[0], z1 = bo[1], z2 = bo[2];
  for (int k = 0; k < 256; ++k) {
    float tv = (float)tr[k];
    z0 += tv * Wo[(size_t)k * 3 + 0];
    z1 += tv * Wo[(size_t)k * 3 + 1];
    z2 += tv * Wo[(size_t)k * 3 + 2];
  }
  float d = fmaxf(sqrtf(z0 * z0 + z1 * z1 + z2 * z2), 1e-6f);
  out[(size_t)r * 3 + 0] = z0 / d;
  out[(size_t)r * 3 + 1] = z1 / d;
  out[(size_t)r * 3 + 2] = z2 / d;
}

// ---------------------------------------------------------------------------
// Host orchestration
// ---------------------------------------------------------------------------
struct WDesc { int idx; int K; int N; int Kp; };

extern "C" void kernel_launch(void* const* d_in, const int* in_sizes, int n_in,
                              void* d_out, int out_size, void* d_ws, size_t ws_size,
                              hipStream_t stream) {
  (void)n_in; (void)out_size; (void)ws_size;
  const int N = in_sizes[0] / 3;   // 16384 vertices
  const int E = in_sizes[2] / 2;   // 131072 edges

  const float* pos   = (const float*)d_in[0];
  const float* nrm   = (const float*)d_in[1];
  const int*   tplEI = (const int*)d_in[2];
  const int*   geoEI = (const int*)d_in[3];
  const int*   batch = (const int*)d_in[4];

  // WMMA weight matrices (d_in index of W; bias follows at idx+1).
  static const WDesc wd[22] = {
    {5, 12, 32, 32},   {7, 32, 32, 32},     // gcu0 tpl
    {9, 12, 32, 32},   {11, 32, 32, 32},    // gcu0 geo
    {13, 64, 64, 64},                       // gcu0 mix
    {15, 128, 128, 128}, {17, 128, 128, 128},
    {19, 128, 128, 128}, {21, 128, 128, 128},
    {23, 256, 256, 256},                    // gcu1
    {25, 512, 256, 512}, {27, 256, 256, 256},
    {29, 512, 256, 512}, {31, 256, 256, 256},
    {33, 512, 512, 512},                    // gcu2
    {35, 832, 1024, 832},                   // glb
    {37, 1862, 1024, 1888}, {39, 1024, 256, 1024},  // heat mlp
    {43, 1862, 1024, 1888}, {45, 1024, 256, 1024},  // skin mlp
    {49, 1862, 1024, 1888}, {51, 1024, 256, 1024},  // conf mlp
  };
  size_t wtoff[22]; size_t wtTotal = 0;
  for (int i = 0; i < 22; ++i) { wtoff[i] = wtTotal; wtTotal += (size_t)wd[i].N * wd[i].Kp; }

  // ---- workspace layout (256B aligned) ----
  char* ws = (char*)d_ws;
  size_t off = 0;
  auto take = [&](size_t bytes) { size_t o = off; off = (off + bytes + 255) & ~(size_t)255; return o; };
  const size_t WT_OFF    = take(wtTotal * 2);
  const size_t X0H_OFF   = take((size_t)N * 8 * 2);
  const size_t EF_OFF    = take((size_t)E * 512 * 2);   // reused later: X4 + V
  const size_t M1_OFF    = take((size_t)E * 256 * 2);   // reused later: H1
  const size_t M2_OFF    = take((size_t)E * 256 * 2);   // reused later: T
  const size_t AGG_OFF   = take((size_t)N * 512 * 4);
  const size_t HCAT_OFF  = take((size_t)N * 512 * 2);
  const size_t FEATS_OFF = take((size_t)N * 832 * 2);
  const size_t XG_OFF    = take((size_t)4 * 1024 * 4);
  const size_t DEN_OFF   = take(96 * 4);

  _Float16* WT    = (_Float16*)(ws + WT_OFF);
  _Float16* X0H   = (_Float16*)(ws + X0H_OFF);
  _Float16* EF    = (_Float16*)(ws + EF_OFF);
  _Float16* M1    = (_Float16*)(ws + M1_OFF);
  _Float16* M2    = (_Float16*)(ws + M2_OFF);
  float*    AGG   = (float*)(ws + AGG_OFF);
  _Float16* HCAT  = (_Float16*)(ws + HCAT_OFF);
  _Float16* FEATS = (_Float16*)(ws + FEATS_OFF);
  float*    XG    = (float*)(ws + XG_OFF);
  float*    DENOM = (float*)(ws + DEN_OFF);
  // aliases (edge buffers are dead by the time these are live)
  float*    X4 = (float*)(ws + EF_OFF);
  _Float16* V  = (_Float16*)(ws + EF_OFF + (size_t)N * 1024 * 4);
  _Float16* H1 = (_Float16*)(ws + M1_OFF);
  _Float16* T  = (_Float16*)(ws + M2_OFF);

  float* out      = (float*)d_out;
  float* coords   = out;                  // (4,24,3)   = 288
  float* hm_out   = out + 288;            // (N,48)
  float* skin_out = out + 288 + (size_t)N * 48;
  float* conf_out = out + 288 + (size_t)N * 48 + (size_t)N * 24;

  auto LB = [&](auto kern, long total, auto... args) {
    long g = (total + 255) / 256;
    kern<<<dim3((unsigned)g), dim3(256), 0, stream>>>(args..., total);
  };
  auto gemm = [&](const _Float16* A, int lda, int wdi, const float* bias,
                  _Float16* oh, float* of, int ldo, int M, int Ncols, int K, int relu) {
    if ((Ncols % 64 == 0) && (M % 128 == 0)) {
      dim3 g(Ncols / 64, M / 128), b(256);
      k_gemm_wmma_2x2<<<g, b, 0, stream>>>(A, lda, WT + wtoff[wdi], bias, oh, of,
                                           ldo, M, Ncols, K, relu);
    } else {
      dim3 g(Ncols / 32, M / 64), b(256);
      k_gemm_wmma<<<g, b, 0, stream>>>(A, lda, WT + wtoff[wdi], bias, oh, of,
                                       ldo, M, Ncols, K, relu);
    }
  };
  auto bptr = [&](int wdi) { return (const float*)d_in[wd[wdi].idx + 1]; };

  // ---- 1. convert/transpose/pad all WMMA weights to fp16 ----
  for (int i = 0; i < 22; ++i)
    LB(k_convw, (long)wd[i].N * wd[i].Kp,
       (const float*)d_in[wd[i].idx], WT + wtoff[i], wd[i].K, wd[i].N, wd[i].Kp);

  // ---- 2. x0 = [pos, normals] (fp16, width 8) ----
  LB(k_build_x0, (long)N * 8, pos, nrm, X0H);

  // ---- 3. three GCU layers ----
  const int ciA[3]  = {6, 64, 256};
  const int coA[3]  = {64, 256, 512};
  const int ldeA[3] = {32, 128, 512};
  const int fOff[3] = {0, 64, 320};
  const int wbase[3] = {0, 5, 10};

  const _Float16* hprev = X0H; int ldh = 8;
  for (int L = 0; L < 3; ++L) {
    const int co = coA[L], coH = co / 2, ci = ciA[L], lde = ldeA[L];
    LB(k_zero_f32, (long)N * co, AGG);
    for (int s = 0; s < 2; ++s) {   // s=0: tpl, s=1: geo
      const int* ei  = s ? geoEI : tplEI;
      const int* src = ei;
      const int* dst = ei + E;
      const int w0 = wbase[L] + 2 * s, w1 = w0 + 1;
      LB(k_edge_gather, (long)E * lde, hprev, ldh, ci, src, dst, EF, lde);
      gemm(EF, lde, w0, bptr(w0), M1, nullptr, coH, E, coH, lde, 1);
      gemm(M1, coH, w1, bptr(w1), M2, nullptr, coH, E, coH, coH, 1);
      LB(k_scatter_max_h, (long)E * coH, M2, coH, dst, AGG, co, s * coH);
    }
    LB(k_f32_to_f16, (long)N * co, AGG, HCAT);
    const int wm = wbase[L] + 4;
    gemm(HCAT, co, wm, bptr(wm), FEATS + fOff[L], nullptr, 832, N, co, co, 1);
    hprev = FEATS + fOff[L]; ldh = 832;
  }

  // ---- 4. glb MLP + batch segment-max ----
  gemm(FEATS, 832, 15, bptr(15), nullptr, X4, 1024, N, 1024, 832, 1);
  LB(k_zero_f32, (long)4 * 1024, XG);
  LB(k_scatter_max_f32, (long)N * 1024, X4, batch, XG);

  // ---- 5. per-vertex feature vector v (N x 1888, padded from 1862) ----
  LB(k_build_v, (long)N * 1888, X0H, FEATS, XG, batch, V);

  // ---- 6. heads ----
  k_init_small<<<1, 512, 0, stream>>>(DENOM, coords);

  // heat
  gemm(V, 1888, 16, bptr(16), H1, nullptr, 1024, N, 1024, 1888, 1);
  gemm(H1, 1024, 17, bptr(17), T, nullptr, 256, N, 256, 1024, 1);
  LB(k_head_heat, (long)N * 48, T, (const float*)d_in[41], (const float*)d_in[42],
     batch, hm_out, DENOM);
  LB(k_coords, (long)N * 24, hm_out, DENOM, pos, batch, coords);

  // skin
  gemm(V, 1888, 18, bptr(18), H1, nullptr, 1024, N, 1024, 1888, 1);
  gemm(H1, 1024, 19, bptr(19), T, nullptr, 256, N, 256, 1024, 1);
  LB(k_linear_out, (long)N * 24, T, (const float*)d_in[47], (const float*)d_in[48],
     skin_out, 24);

  // conf
  gemm(V, 1888, 20, bptr(20), H1, nullptr, 1024, N, 1024, 1888, 1);
  gemm(H1, 1024, 21, bptr(21), T, nullptr, 256, N, 256, 1024, 1);
  k_conf<<<dim3((N + 255) / 256), dim3(256), 0, stream>>>(
      T, (const float*)d_in[53], (const float*)d_in[54], conf_out, N);
}
